// MultiHeadAttention_55774445306043
// MI455X (gfx1250) — compile-verified, matrix operand order checked by reading, not run on hardware
//
#include <hip/hip_runtime.h>
#include <hip/hip_bf16.h>
#include <stdint.h>

// ---------------------------------------------------------------------------
// MultiHeadAttention, B=1, N=16384, C=256, H=2, Dh=128, fp32 in/out.
// bf16 WMMA everywhere; flash attention with double-buffered async-to-LDS
// K/V staging (GLOBAL_LOAD_ASYNC_TO_LDS_B128 + s_wait_asynccnt).
// ---------------------------------------------------------------------------

typedef __attribute__((ext_vector_type(16))) __bf16 v16bf;
typedef __attribute__((ext_vector_type(8)))  float  v8f;
typedef int vi4 __attribute__((vector_size(16)));   // matches builtin proto

#define N_TOK   16384
#define C_DIM   256
#define QKV_DIM 768
#define DH      128
#define BN      32          // keys per flash block
#define NBLK    (N_TOK / BN)

#if defined(__HIP_DEVICE_COMPILE__) && __has_builtin(__builtin_amdgcn_global_load_async_to_lds_b128)
#define HAVE_ASYNC_LDS 1
#else
#define HAVE_ASYNC_LDS 0
#endif

union BF16x16 { v16bf v; uint4 u[2]; };

__device__ __forceinline__ v8f wmma_bf16(v16bf a, v16bf b, v8f c) {
  return __builtin_amdgcn_wmma_f32_16x16x32_bf16(
      /*neg_a=*/false, a, /*neg_b=*/false, b,
      /*c_mod=*/(short)0, c, /*reuse_a=*/false, /*reuse_b=*/false);
}

// 16-byte global -> LDS copy; async (ASYNCcnt-tracked) when available.
__device__ __forceinline__ void copy16_g2l(const __bf16* gsrc, __bf16* lds) {
#if HAVE_ASYNC_LDS
  __builtin_amdgcn_global_load_async_to_lds_b128(
      (vi4*)gsrc, (vi4*)lds, /*imm offset=*/0, /*cpol=*/0);
#else
  *(uint4*)lds = *(const uint4*)gsrc;
#endif
}

__device__ __forceinline__ void wait_async_lds() {
#if HAVE_ASYNC_LDS
#if __has_builtin(__builtin_amdgcn_s_wait_asynccnt)
  __builtin_amdgcn_s_wait_asynccnt(0);
#else
  asm volatile("s_wait_asynccnt 0" ::: "memory");
#endif
#endif
}

// ---------------------------------------------------------------------------
// Kernel 1: h = x + pos ; QKV = h @ W_qkv^T   -> bf16 [N][768]
// ---------------------------------------------------------------------------
__global__ __launch_bounds__(256) void qkv_kernel(
    const float* __restrict__ x, const float* __restrict__ pos,
    const float* __restrict__ Wqkv, __bf16* __restrict__ qkv) {
  const int wave  = threadIdx.x >> 5;
  const int lane  = threadIdx.x & 31;
  const int l16   = lane & 15;
  const int lhalf = lane >> 4;
  const int rowBase = blockIdx.x * 128 + wave * 16;
  const int colBase = blockIdx.y * 16;

  const float* xr = x   + (size_t)(rowBase + l16) * C_DIM;
  const float* pr = pos + (size_t)(rowBase + l16) * C_DIM;
  const float* wr = Wqkv + (size_t)(colBase + l16) * C_DIM;

  v8f acc = {};
#pragma unroll
  for (int kb = 0; kb < C_DIM / 32; ++kb) {
    const int k0 = kb * 32 + lhalf * 8;       // a[0..7], b[0..7]
    const int k1 = k0 + 16;                   // a[8..15], b[8..15]
    v16bf a, b;
#pragma unroll
    for (int j = 0; j < 8; ++j) {
      a[j]     = (__bf16)(xr[k0 + j] + pr[k0 + j]);
      a[8 + j] = (__bf16)(xr[k1 + j] + pr[k1 + j]);
      b[j]     = (__bf16)wr[k0 + j];
      b[8 + j] = (__bf16)wr[k1 + j];
    }
    acc = wmma_bf16(a, b, acc);
  }
#pragma unroll
  for (int v = 0; v < 8; ++v) {
    const int orow = rowBase + v + 8 * lhalf;
    qkv[(size_t)orow * QKV_DIM + colBase + l16] = (__bf16)acc[v];
  }
}

// ---------------------------------------------------------------------------
// Kernel 2: flash attention per head. grid (128, 2), block 256 (8 waves).
// Double-buffered LDS staging: tile j+1 streams in (async) while tile j
// feeds the WMMAs.
// ---------------------------------------------------------------------------
__global__ __launch_bounds__(256) void attn_kernel(
    const __bf16* __restrict__ qkv, __bf16* __restrict__ atto) {
  __shared__ __align__(16) __bf16 Ks[2][BN * DH];      // [m][c]  2 x 8KB
  __shared__ __align__(16) __bf16 VsT[2][DH * BN];     // [c][m]  2 x 8KB
  __shared__ __align__(16) __bf16 Ps[8 * 16 * BN];     // per-wave P, 8KB

  const int tid   = threadIdx.x;
  const int wave  = tid >> 5;
  const int lane  = tid & 31;
  const int l16   = lane & 15;
  const int lhalf = lane >> 4;
  const int head  = blockIdx.y;
  const int qBase = blockIdx.x * 128 + wave * 16;
  const float scale = 0.088388347648318447f;   // 1/sqrt(128)

  // ---- stage K (async b128) and V^T (register transpose) for tile j ----
  auto stage = [&](int j, int buf) {
    const int kb0 = j * BN;
#pragma unroll
    for (int p = 0; p < 2; ++p) {
      const int idx = tid + p * 256;     // 512 segments of 8 bf16
      const int r = idx >> 4;            // key row 0..31
      const int s = idx & 15;            // 8-elem segment along c
      const size_t rowoff = (size_t)(kb0 + r) * QKV_DIM + head * DH;
      copy16_g2l(qkv + rowoff + C_DIM + s * 8, &Ks[buf][r * DH + s * 8]);
      uint4 vv = *((const uint4*)(qkv + rowoff + 2 * C_DIM) + s);
      const __bf16* ve = (const __bf16*)&vv;
#pragma unroll
      for (int e = 0; e < 8; ++e) VsT[buf][(s * 8 + e) * BN + r] = ve[e];
    }
  };

  // ---- load Q fragments (A-layout) for this wave's 16 rows ----
  v16bf qf[DH / 32];
  {
    const __bf16* qrow = qkv + (size_t)(qBase + l16) * QKV_DIM + head * DH;
#pragma unroll
    for (int kb = 0; kb < DH / 32; ++kb) {
      BF16x16 f;
      f.u[0] = *(const uint4*)(qrow + kb * 32 + lhalf * 8);
      f.u[1] = *(const uint4*)(qrow + kb * 32 + 16 + lhalf * 8);
      qf[kb] = f.v;
    }
  }

  v8f o[DH / 16];
  v8f vzero = {};
#pragma unroll
  for (int i = 0; i < DH / 16; ++i) o[i] = vzero;
  float mrow[8], lrow[8];
#pragma unroll
  for (int v = 0; v < 8; ++v) { mrow[v] = -3.0e38f; lrow[v] = 0.f; }

  stage(0, 0);
  wait_async_lds();
  __syncthreads();

  for (int j = 0; j < NBLK; ++j) {
    const int buf = j & 1;

    // ---- prefetch pipeline: stage j+1, L2-prefetch j+2 ----
    if (j + 1 < NBLK) {
      stage(j + 1, buf ^ 1);
      if (j + 2 < NBLK) {
        const int r = lane;
        const size_t rowoff =
            (size_t)((j + 2) * BN + r) * QKV_DIM + head * DH;
        __builtin_prefetch(qkv + rowoff + C_DIM, 0, 1);     // K line
        __builtin_prefetch(qkv + rowoff + 2 * C_DIM, 0, 1); // V line
      }
    }

    // ---- S = Q @ K^T (two 16-key column chunks) ----
    v8f s0 = vzero, s1 = vzero;
#pragma unroll
    for (int kb = 0; kb < DH / 32; ++kb) {
      const __bf16* kr0 = &Ks[buf][(size_t)l16 * DH];
      const __bf16* kr1 = &Ks[buf][(size_t)(16 + l16) * DH];
      BF16x16 b0, b1;
      b0.u[0] = *(const uint4*)(kr0 + kb * 32 + lhalf * 8);
      b0.u[1] = *(const uint4*)(kr0 + kb * 32 + 16 + lhalf * 8);
      b1.u[0] = *(const uint4*)(kr1 + kb * 32 + lhalf * 8);
      b1.u[1] = *(const uint4*)(kr1 + kb * 32 + 16 + lhalf * 8);
      s0 = wmma_bf16(qf[kb], b0.v, s0);
      s1 = wmma_bf16(qf[kb], b1.v, s1);
    }

    // ---- online softmax (rows in VGPR index; reduce across 16 lanes) ----
    float alpha[8];
#pragma unroll
    for (int v = 0; v < 8; ++v) {
      const float a0 = s0[v] * scale, a1 = s1[v] * scale;
      float m = fmaxf(a0, a1);
      m = fmaxf(m, __shfl_xor(m, 1, 32));
      m = fmaxf(m, __shfl_xor(m, 2, 32));
      m = fmaxf(m, __shfl_xor(m, 4, 32));
      m = fmaxf(m, __shfl_xor(m, 8, 32));
      const float mnew = fmaxf(mrow[v], m);
      alpha[v] = __expf(mrow[v] - mnew);
      mrow[v] = mnew;
      const float p0 = __expf(a0 - mnew);
      const float p1 = __expf(a1 - mnew);
      s0[v] = p0; s1[v] = p1;
      float ps = p0 + p1;
      ps += __shfl_xor(ps, 1, 32);
      ps += __shfl_xor(ps, 2, 32);
      ps += __shfl_xor(ps, 4, 32);
      ps += __shfl_xor(ps, 8, 32);
      lrow[v] = lrow[v] * alpha[v] + ps;
    }
#pragma unroll
    for (int of = 0; of < DH / 16; ++of)
#pragma unroll
      for (int v = 0; v < 8; ++v) o[of][v] *= alpha[v];

    // ---- re-layout P (C-frag -> A-frag) via per-wave LDS tile ----
    __bf16* pw = Ps + wave * 16 * BN;
#pragma unroll
    for (int v = 0; v < 8; ++v) {
      const int row = v + 8 * lhalf;
      pw[row * BN + l16]      = (__bf16)s0[v];
      pw[row * BN + 16 + l16] = (__bf16)s1[v];
    }
    BF16x16 pf;
    pf.u[0] = *(const uint4*)(pw + l16 * BN + lhalf * 8);
    pf.u[1] = *(const uint4*)(pw + l16 * BN + 16 + lhalf * 8);

    // ---- O += P @ V  (V^T staged, so B frags are contiguous b128 reads) ----
#pragma unroll
    for (int of = 0; of < DH / 16; ++of) {
      const __bf16* vc = &VsT[buf][(size_t)(of * 16 + l16) * BN];
      BF16x16 bv;
      bv.u[0] = *(const uint4*)(vc + lhalf * 8);
      bv.u[1] = *(const uint4*)(vc + 16 + lhalf * 8);
      o[of] = wmma_bf16(pf.v, bv.v, o[of]);
    }

    // publish tile j+1 (async K writes + V ds writes) to all waves
    wait_async_lds();
    __syncthreads();
  }

  // ---- normalize and store bf16 attention output [N][C] ----
  float rinv[8];
#pragma unroll
  for (int v = 0; v < 8; ++v) rinv[v] = 1.0f / lrow[v];
#pragma unroll
  for (int of = 0; of < DH / 16; ++of)
#pragma unroll
    for (int v = 0; v < 8; ++v) {
      const int row = qBase + v + 8 * lhalf;
      atto[(size_t)row * C_DIM + head * DH + of * 16 + l16] =
          (__bf16)(o[of][v] * rinv[v]);
    }
}

// ---------------------------------------------------------------------------
// Kernel 3: out = atto @ W_proj^T + b_proj  (fp32 out)
// ---------------------------------------------------------------------------
__global__ __launch_bounds__(256) void proj_kernel(
    const __bf16* __restrict__ atto, const float* __restrict__ Wproj,
    const float* __restrict__ bproj, float* __restrict__ out) {
  const int wave  = threadIdx.x >> 5;
  const int lane  = threadIdx.x & 31;
  const int l16   = lane & 15;
  const int lhalf = lane >> 4;
  const int rowBase = blockIdx.x * 128 + wave * 16;
  const int colBase = blockIdx.y * 16;

  const __bf16* ar = atto + (size_t)(rowBase + l16) * C_DIM;
  const float*  wr = Wproj + (size_t)(colBase + l16) * C_DIM;

  v8f acc = {};
#pragma unroll
  for (int kb = 0; kb < C_DIM / 32; ++kb) {
    const int k0 = kb * 32 + lhalf * 8;
    const int k1 = k0 + 16;
    BF16x16 a;
    a.u[0] = *(const uint4*)(ar + k0);
    a.u[1] = *(const uint4*)(ar + k1);
    v16bf b;
#pragma unroll
    for (int j = 0; j < 8; ++j) {
      b[j]     = (__bf16)wr[k0 + j];
      b[8 + j] = (__bf16)wr[k1 + j];
    }
    acc = wmma_bf16(a.v, b, acc);
  }
  const float bias = bproj[colBase + l16];
#pragma unroll
  for (int v = 0; v < 8; ++v) {
    const int orow = rowBase + v + 8 * lhalf;
    out[(size_t)orow * C_DIM + colBase + l16] = acc[v] + bias;
  }
}

// ---------------------------------------------------------------------------
extern "C" void kernel_launch(void* const* d_in, const int* in_sizes, int n_in,
                              void* d_out, int out_size, void* d_ws, size_t ws_size,
                              hipStream_t stream) {
  const float* x     = (const float*)d_in[0];
  const float* pos   = (const float*)d_in[1];
  const float* Wqkv  = (const float*)d_in[2];
  const float* Wproj = (const float*)d_in[3];
  const float* bproj = (const float*)d_in[4];
  float* out = (float*)d_out;

  __bf16* qkv  = (__bf16*)d_ws;                                      // 24 MB
  __bf16* atto = (__bf16*)((char*)d_ws +
                           (size_t)N_TOK * QKV_DIM * sizeof(__bf16)); // +8 MB

  dim3 g1(N_TOK / 128, QKV_DIM / 16);
  qkv_kernel<<<g1, 256, 0, stream>>>(x, pos, Wqkv, qkv);

  dim3 g2(N_TOK / 128, 2);
  attn_kernel<<<g2, 256, 0, stream>>>(qkv, atto);

  dim3 g3(N_TOK / 128, C_DIM / 16);
  proj_kernel<<<g3, 256, 0, stream>>>(atto, Wproj, bproj, out);
}
